// FP8Linear_43078521978973
// MI455X (gfx1250) — 8-bit matrix operand order repaired and checked by reading, compiled, not run on hardware
//
#include <hip/hip_runtime.h>
#include <hip/hip_fp16.h>

#define MDIM 32
#define NDIM 8192
#define KDIM 8192
#define KSLICES 4
#define KSPAN (KDIM / KSLICES)
#define KITER (KSPAN / 128)
#define NT 2
#define SCALE_B 0.05f
#define LO_SCALE 128.0f
#define ASIZE (MDIM * KDIM)

typedef __attribute__((ext_vector_type(16))) int          v16i;
typedef __attribute__((ext_vector_type(8)))  float        v8f;
typedef __attribute__((ext_vector_type(4))) unsigned int  v4u;


__device__ __forceinline__ float e4m3_decode(unsigned v) {
  unsigned e = (v >> 3) & 0xFu;
  unsigned m = v & 7u;
  float mag = (e == 0u) ? (float)m * 0.001953125f
                        : __uint_as_float(((e + 120u) << 23) | (m << 20));
  return (v & 0x80u) ? -mag : mag;
}

__device__ __forceinline__ unsigned e4m3_encode(float x) {
  unsigned u  = __float_as_uint(x);
  unsigned s  = (u >> 24) & 0x80u;
  unsigned au = u & 0x7FFFFFFFu;
  float ax = __uint_as_float(au);
  if (ax < 0.0009765625f) return s;
  if (ax >= 448.0f)       return s | 0x7Eu;
  int e32 = (int)(au >> 23) - 127;
  if (e32 >= -6) {
    unsigned r = au + 0x7FFFFu + ((au >> 20) & 1u);
    int e = (int)(r >> 23) - 127;
    if (e > 8) return s | 0x7Eu;
    return s | ((unsigned)(e + 7) << 3) | ((r >> 20) & 7u);
  }
  int n = (int)rintf(ax * 512.0f);
  if (n >= 8) return s | 0x08u;
  return s | (unsigned)n;
}

__global__ void prep_a_kernel(const __half* __restrict__ A,
                              unsigned char* __restrict__ a8p) {
  int i = blockIdx.x * blockDim.x + threadIdx.x;
  if (i >= MDIM * KDIM) return;
  float a = __half2float(A[i]);
  unsigned hi = e4m3_encode(a);
  float r = (a - e4m3_decode(hi)) * LO_SCALE;
  unsigned lo = e4m3_encode(r);

  const int m = i >> 13;
  const int k = i & (KDIM - 1);
  const int mt = m >> 4;
  const int l16 = m & 15;
  const int kb = k >> 7;
  const int h  = (k >> 4) & 1;
  const int lane = h * 16 + l16;
  const int off  = ((k >> 5) & 3) * 16 + (k & 15);
  const size_t base = ((size_t)(kb * 2 + mt) * 32 + lane) * 64 + off;
  a8p[base]         = (unsigned char)hi;
  a8p[base + ASIZE] = (unsigned char)lo;
}

__global__ void __launch_bounds__(256, 1)
fp8_gemm_kernel(const unsigned char* __restrict__ a8p,
                const unsigned char* __restrict__ B,
                float* __restrict__ partial) {
  const int lane = threadIdx.x & 31;
  const int wave = (int)blockIdx.x * 8 + (threadIdx.x >> 5);
  const int ks   = wave >> 8;
  const int nw   = wave & 255;
  const int n0   = nw * (NT * 16);
  const int h    = lane >> 4;
  const int l16  = lane & 15;

  v8f acc_hi[2][NT], acc_lo[2][NT];
#pragma unroll
  for (int mt = 0; mt < 2; ++mt)
#pragma unroll
    for (int nt = 0; nt < NT; ++nt) {
      acc_hi[mt][nt] = (v8f){0.f,0.f,0.f,0.f,0.f,0.f,0.f,0.f};
      acc_lo[mt][nt] = (v8f){0.f,0.f,0.f,0.f,0.f,0.f,0.f,0.f};
    }

  const int kbeg = ks * KSPAN;

  const unsigned char* __restrict__ paw =
      a8p + (size_t)(kbeg >> 7) * 4096 + (size_t)lane * 64;
  const unsigned char* __restrict__ pbw =
      B + (size_t)(n0 + l16) * KDIM + kbeg + h * 16;

  v16i bf[2][NT];
#pragma unroll
  for (int nt = 0; nt < NT; ++nt) {
    const unsigned char* pb = pbw + nt * (16 * KDIM);
#pragma unroll
    for (int g = 0; g < 4; ++g) {
      v4u t = __builtin_nontemporal_load((const v4u*)(pb + g * 32));
      bf[0][nt][g * 4 + 0] = (int)t.x; bf[0][nt][g * 4 + 1] = (int)t.y;
      bf[0][nt][g * 4 + 2] = (int)t.z; bf[0][nt][g * 4 + 3] = (int)t.w;
    }
  }

#pragma unroll
  for (int it = 0; it < KITER; ++it) {
    const int cur = it & 1;
    const int nxt = cur ^ 1;

    v16i ah[2], al[2];
#pragma unroll
    for (int mt = 0; mt < 2; ++mt) {
      ah[mt] = *(const v16i*)(paw + it * 4096 + mt * 2048);
      al[mt] = *(const v16i*)(paw + it * 4096 + mt * 2048 + ASIZE);
    }

    if (it + 1 < KITER) {
#pragma unroll
      for (int nt = 0; nt < NT; ++nt) {
        const unsigned char* pb = pbw + nt * (16 * KDIM) + (it + 1) * 128;
        __builtin_prefetch(pb + 384, 0, 0);
#pragma unroll
        for (int g = 0; g < 4; ++g) {
          v4u t = __builtin_nontemporal_load((const v4u*)(pb + g * 32));
          bf[nxt][nt][g * 4 + 0] = (int)t.x; bf[nxt][nt][g * 4 + 1] = (int)t.y;
          bf[nxt][nt][g * 4 + 2] = (int)t.z; bf[nxt][nt][g * 4 + 3] = (int)t.w;
        }
      }
    }

#pragma unroll
    for (int nt = 0; nt < NT; ++nt)
#pragma unroll
      for (int mt = 0; mt < 2; ++mt) {
        acc_hi[mt][nt] = __builtin_amdgcn_wmma_f32_16x16x128_fp8_fp8(
            ah[mt], bf[cur][nt], (short)0, acc_hi[mt][nt], false, false);
        acc_lo[mt][nt] = __builtin_amdgcn_wmma_f32_16x16x128_fp8_fp8(
            al[mt], bf[cur][nt], (short)0, acc_lo[mt][nt], false, false);
      }
  }

  float* __restrict__ p = partial + (size_t)ks * (MDIM * NDIM);
#pragma unroll
  for (int mt = 0; mt < 2; ++mt)
#pragma unroll
    for (int nt = 0; nt < NT; ++nt) {
      const int n = n0 + nt * 16 + l16;
#pragma unroll
      for (int r = 0; r < 8; ++r) {
        const int m = mt * 16 + h * 8 + r;
        p[m * NDIM + n] =
            acc_hi[mt][nt][r] + acc_lo[mt][nt][r] * (1.0f / LO_SCALE);
      }
    }
}

__global__ void combine_kernel(const float* __restrict__ partial,
                               float* __restrict__ out) {
  int i = blockIdx.x * blockDim.x + threadIdx.x;
  if (i >= MDIM * NDIM) return;
  float s = 0.f;
#pragma unroll
  for (int k = 0; k < KSLICES; ++k) s += partial[(size_t)k * (MDIM * NDIM) + i];
  out[i] = SCALE_B * s;
}

extern "C" void kernel_launch(void* const* d_in, const int* in_sizes, int n_in,
                              void* d_out, int out_size, void* d_ws, size_t ws_size,
                              hipStream_t stream) {
  (void)in_sizes; (void)n_in; (void)out_size; (void)ws_size;
  const __half* A          = (const __half*)d_in[0];
  const unsigned char* B   = (const unsigned char*)d_in[1];
  float* out               = (float*)d_out;
  unsigned char* a8p       = (unsigned char*)d_ws;
  float* partial = (float*)((char*)d_ws + (size_t)(2 * ASIZE));

  prep_a_kernel<<<(MDIM * KDIM + 255) / 256, 256, 0, stream>>>(A, a8p);

  const int total_waves = KSLICES * (NDIM / (NT * 16));
  fp8_gemm_kernel<<<total_waves * 32 / 256, 256, 0, stream>>>(a8p, B, partial);

  combine_kernel<<<(MDIM * NDIM + 255) / 256, 256, 0, stream>>>(partial, out);
}
